// sequence_2_sequence_LSTM_90228672955269
// MI455X (gfx1250) — matrix operand order repaired and checked by reading, not compiled, not run on hardware
//
#include <hip/hip_runtime.h>
#include <hip/hip_bf16.h>

// ---------------------------------------------------------------------------
// Types for CDNA5 WMMA (wave32): bf16 A/B fragments (16 VGPR halves), f32 accum
// ---------------------------------------------------------------------------
typedef __attribute__((ext_vector_type(16))) __bf16 v16bf;
typedef __attribute__((ext_vector_type(8)))  float  v8f;

#define HDIM   1024
#define BATCH  128
#define TENC   32
#define LDEC   20
#define FIN    2048
#define WVDIM  300
#define WVPAD  320      // decoder x-part K padded to multiple of 32
#define KENC   (FIN + HDIM)    // 3072
#define KDEC   (WVPAD + HDIM)  // 1344
#define NGATE  (4 * HDIM)      // 4096
#define NPROJ  304      // 300 padded to 19 tiles of 16

__device__ __forceinline__ unsigned short f2bf(float x) {
  unsigned int u = __float_as_uint(x);
  u = (u + 0x7FFFu + ((u >> 16) & 1u)) >> 16;   // round-to-nearest-even
  return (unsigned short)u;
}

__device__ __forceinline__ float sigmoidf_(float x) {
  return 1.0f / (1.0f + __expf(-x));
}

// A fragment 16x32 bf16, row-major source. Per ISA: lane L<16 holds row M=L,
// K = {kb..kb+7, kb+16..kb+23}; lane L+16 holds row M=L, K = {kb+8.., kb+24..}.
__device__ __forceinline__ v16bf load_fragA(const unsigned short* __restrict__ base,
                                            int row_stride, int m0, int kb, int lane) {
  int r = m0 + (lane & 15);
  int koff = kb + ((lane >> 4) << 3);
  const unsigned short* p = base + (long)r * row_stride + koff;
  union { uint4 q[2]; v16bf v; } f;
  f.q[0] = *(const uint4*)(p);        // K koff..koff+7
  f.q[1] = *(const uint4*)(p + 16);   // K koff+16..koff+23
  return f.v;
}

// B fragment 32x16 bf16 from weights pre-transposed to [N][K]. Per ISA B layout:
// lane L<16 holds col N=L, K = {kb..kb+7, kb+16..kb+23}; lane L+16 holds K = {kb+8.., kb+24..}.
__device__ __forceinline__ v16bf load_fragB(const unsigned short* __restrict__ Wt,
                                            int Ktot, int n, int kb, int lane) {
  const unsigned short* p = Wt + (long)n * Ktot + kb + ((lane >> 4) << 3);
  union { uint4 q[2]; v16bf v; } f;
  f.q[0] = *(const uint4*)(p);
  f.q[1] = *(const uint4*)(p + 16);
  return f.v;
}

// ---------------------------------------------------------------------------
// One LSTM step:  z = [x, h] @ W + b ; gate math ; c (in place), h_out (bf16).
// Wave = 2 M-tiles (32 batch rows) x 16 H columns x all 4 gates:
//   - A fragments reused across 4 gates
//   - B fragments reused across both M-tiles (register-level reuse)
//   -> 12 b128 loads per 8 WMMAs per k-step (1.5 loads/WMMA)
// Grid: 4 m-groups * 64 n-tiles = 256 waves = 32 blocks of 256 threads.
// ---------------------------------------------------------------------------
__global__ __launch_bounds__(256) void lstm_gates_kernel(
    const unsigned short* __restrict__ Ax, int ax_stride, int Kx,
    const unsigned short* __restrict__ Ah,                 // [B, H]
    const unsigned short* __restrict__ Wt, int Ktot,       // [4H, Ktot] bf16, transposed
    const float* __restrict__ bias,                        // [4H]
    float* __restrict__ c,                                 // [B, H] in/out
    unsigned short* __restrict__ h_out)                    // [B, H] bf16
{
  int wave = (blockIdx.x * blockDim.x + threadIdx.x) >> 5;
  int lane = threadIdx.x & 31;
  int n_tile = wave & 63;          // 64 tiles over H
  int m0 = (wave >> 6) << 5;       // 4 m-groups of 32 rows
  int nh0 = n_tile << 4;

  v8f acc[2][4];                    // [m-tile][gate i,j,f,o]
#pragma unroll
  for (int mt = 0; mt < 2; ++mt)
#pragma unroll
    for (int g = 0; g < 4; ++g) acc[mt][g] = v8f{};

  // x-part of K
  for (int kb = 0; kb < Kx; kb += 32) {
    v16bf a0 = load_fragA(Ax, ax_stride, m0,      kb, lane);
    v16bf a1 = load_fragA(Ax, ax_stride, m0 + 16, kb, lane);
#pragma unroll
    for (int g = 0; g < 4; ++g) {
      v16bf b = load_fragB(Wt, Ktot, nh0 + g * HDIM + (lane & 15), kb, lane);
      acc[0][g] = __builtin_amdgcn_wmma_f32_16x16x32_bf16(
          false, a0, false, b, (short)0, acc[0][g], false, false);
      acc[1][g] = __builtin_amdgcn_wmma_f32_16x16x32_bf16(
          false, a1, false, b, (short)0, acc[1][g], false, false);
    }
  }
  // h-part of K
  for (int kb = 0; kb < HDIM; kb += 32) {
    v16bf a0 = load_fragA(Ah, HDIM, m0,      kb, lane);
    v16bf a1 = load_fragA(Ah, HDIM, m0 + 16, kb, lane);
#pragma unroll
    for (int g = 0; g < 4; ++g) {
      v16bf b = load_fragB(Wt, Ktot, nh0 + g * HDIM + (lane & 15), Kx + kb, lane);
      acc[0][g] = __builtin_amdgcn_wmma_f32_16x16x32_bf16(
          false, a0, false, b, (short)0, acc[0][g], false, false);
      acc[1][g] = __builtin_amdgcn_wmma_f32_16x16x32_bf16(
          false, a1, false, b, (short)0, acc[1][g], false, false);
    }
  }

  // Fused gate math in the C/D layout: element r of lane L is
  // (M = m_base + r + 8*(L>>4), N = nh0 + (L&15)).
  int ncol = nh0 + (lane & 15);
  float bi = bias[ncol];
  float bj = bias[ncol + HDIM];
  float bf = bias[ncol + 2 * HDIM];
  float bo = bias[ncol + 3 * HDIM];
#pragma unroll
  for (int mt = 0; mt < 2; ++mt) {
    int m_base = m0 + mt * 16 + ((lane >> 4) << 3);
#pragma unroll
    for (int r = 0; r < 8; ++r) {
      int m = m_base + r;
      float zi = acc[mt][0][r] + bi;
      float zj = acc[mt][1][r] + bj;
      float zf = acc[mt][2][r] + bf + 1.0f;     // FORGET_BIAS
      float zo = acc[mt][3][r] + bo;
      float cn = c[m * HDIM + ncol] * sigmoidf_(zf) + sigmoidf_(zi) * tanhf(zj);
      c[m * HDIM + ncol] = cn;
      h_out[m * HDIM + ncol] = f2bf(tanhf(cn) * sigmoidf_(zo));
    }
  }
}

// ---------------------------------------------------------------------------
// Projection: [L*B, 1024] @ projW[1024, 300] + b -> out [B, L, 300] (fp32)
// ---------------------------------------------------------------------------
__global__ __launch_bounds__(256) void proj_kernel(
    const unsigned short* __restrict__ hs,   // [L*B, H] bf16
    const unsigned short* __restrict__ Wt,   // [NPROJ, H] bf16 transposed, padded
    const float* __restrict__ pb,            // [300]
    float* __restrict__ out)                 // [B, L, 300]
{
  const int M_TILES = (LDEC * BATCH) / 16;   // 160
  const int N_TILES = NPROJ / 16;            // 19
  int wave = (blockIdx.x * blockDim.x + threadIdx.x) >> 5;
  if (wave >= M_TILES * N_TILES) return;
  int lane = threadIdx.x & 31;
  int n0 = (wave % N_TILES) << 4;
  int m0 = (wave / N_TILES) << 4;

  v8f acc = v8f{};
  for (int kb = 0; kb < HDIM; kb += 32) {
    v16bf a = load_fragA(hs, HDIM, m0, kb, lane);
    v16bf b = load_fragB(Wt, HDIM, n0 + (lane & 15), kb, lane);
    acc = __builtin_amdgcn_wmma_f32_16x16x32_bf16(
        false, a, false, b, (short)0, acc, false, false);
  }

  int n = n0 + (lane & 15);
  if (n < WVDIM) {
    float bv = pb[n];
#pragma unroll
    for (int r = 0; r < 8; ++r) {
      int m = m0 + r + ((lane >> 4) << 3);   // m = l*BATCH + b
      int l = m >> 7;
      int b = m & (BATCH - 1);
      out[(b * LDEC + l) * WVDIM + n] = acc[r] + bv;
    }
  }
}

// ---------------------------------------------------------------------------
// One-time conversion / transpose kernels (fp32 -> bf16)
// ---------------------------------------------------------------------------
__global__ void transpose_bf16_kernel(const float* __restrict__ in,  // [K, N]
                                      unsigned short* __restrict__ out, // [N, K]
                                      int K, int N) {
  long idx = (long)blockIdx.x * blockDim.x + threadIdx.x;
  if (idx >= (long)K * N) return;
  int n = (int)(idx / K);
  int k = (int)(idx % K);
  out[idx] = f2bf(in[(long)k * N + n]);
}

// dec_kernel [1324, 4096] -> [4096, 1344] with x-part K padded 300 -> 320
__global__ void transpose_dec_kernel(const float* __restrict__ in,
                                     unsigned short* __restrict__ out) {
  long idx = (long)blockIdx.x * blockDim.x + threadIdx.x;
  if (idx >= (long)NGATE * KDEC) return;
  int n = (int)(idx / KDEC);
  int k = (int)(idx % KDEC);
  float v = 0.0f;
  if (k < WVDIM)       v = in[(long)k * NGATE + n];
  else if (k >= WVPAD) v = in[(long)(WVDIM + k - WVPAD) * NGATE + n];
  out[idx] = f2bf(v);
}

// proj_W [1024, 300] -> [NPROJ, 1024], rows 300..303 zero
__global__ void transpose_proj_kernel(const float* __restrict__ in,
                                      unsigned short* __restrict__ out) {
  long idx = (long)blockIdx.x * blockDim.x + threadIdx.x;
  if (idx >= (long)NPROJ * HDIM) return;
  int n = (int)(idx / HDIM);
  int k = (int)(idx % HDIM);
  out[idx] = f2bf(n < WVDIM ? in[(long)k * WVDIM + n] : 0.0f);
}

__global__ void convert_bf16_kernel(const float* __restrict__ in,
                                    unsigned short* __restrict__ out, long n) {
  long idx = (long)blockIdx.x * blockDim.x + threadIdx.x;
  if (idx < n) out[idx] = f2bf(in[idx]);
}

// Gather caption embeddings for step t into padded bf16 buffer [B, WVPAD]
__global__ void embed_gather_kernel(const int* __restrict__ caption,  // [B, L]
                                    int t,
                                    const float* __restrict__ emb,    // [V, 300]
                                    unsigned short* __restrict__ out) // [B, 320]
{
  int idx = blockIdx.x * blockDim.x + threadIdx.x;
  if (idx >= BATCH * WVPAD) return;
  int b = idx / WVPAD;
  int k = idx % WVPAD;
  float v = 0.0f;
  if (k < WVDIM) v = emb[(long)caption[b * LDEC + t] * WVDIM + k];
  out[idx] = f2bf(v);
}

// ---------------------------------------------------------------------------
// Host-side orchestration (all on `stream`, graph-capture safe)
// ---------------------------------------------------------------------------
extern "C" void kernel_launch(void* const* d_in, const int* in_sizes, int n_in,
                              void* d_out, int out_size, void* d_ws, size_t ws_size,
                              hipStream_t stream) {
  const float* frames     = (const float*)d_in[0];  // [B, T, F]
  const int*   caption    = (const int*)d_in[1];    // [B, L]
  const float* embeddings = (const float*)d_in[2];  // [V, 300]
  const float* enc_kernel = (const float*)d_in[3];  // [3072, 4096]
  const float* enc_bias   = (const float*)d_in[4];  // [4096]
  const float* dec_kernel = (const float*)d_in[5];  // [1324, 4096]
  const float* dec_bias   = (const float*)d_in[6];  // [4096]
  const float* proj_W     = (const float*)d_in[7];  // [1024, 300]
  const float* proj_b     = (const float*)d_in[8];  // [300]
  float* out = (float*)d_out;

  // Workspace layout (all sizes 256B-aligned)
  char* ws = (char*)d_ws;
  size_t off = 0;
  unsigned short* encWt = (unsigned short*)(ws + off); off += (size_t)NGATE * KENC * 2;  // 25.2 MB
  unsigned short* decWt = (unsigned short*)(ws + off); off += (size_t)NGATE * KDEC * 2;  // 11.0 MB
  unsigned short* prjWt = (unsigned short*)(ws + off); off += (size_t)NPROJ * HDIM * 2;  // 0.62 MB
  unsigned short* frBf  = (unsigned short*)(ws + off); off += (size_t)BATCH * TENC * FIN * 2; // 16.8 MB
  float*          cBuf  = (float*)(ws + off);          off += (size_t)BATCH * HDIM * 4;  // 0.5 MB
  unsigned short* h0    = (unsigned short*)(ws + off); off += (size_t)BATCH * HDIM * 2;
  unsigned short* h1    = (unsigned short*)(ws + off); off += (size_t)BATCH * HDIM * 2;
  unsigned short* embBf = (unsigned short*)(ws + off); off += (size_t)BATCH * WVPAD * 2;
  unsigned short* hsBuf = (unsigned short*)(ws + off); off += (size_t)LDEC * BATCH * HDIM * 2; // 5.2 MB

  // One-time weight transposition + precision conversion (L2-resident afterwards)
  {
    long n = (long)NGATE * KENC;
    transpose_bf16_kernel<<<(unsigned)((n + 255) / 256), 256, 0, stream>>>(enc_kernel, encWt, KENC, NGATE);
  }
  {
    long n = (long)NGATE * KDEC;
    transpose_dec_kernel<<<(unsigned)((n + 255) / 256), 256, 0, stream>>>(dec_kernel, decWt);
  }
  {
    long n = (long)NPROJ * HDIM;
    transpose_proj_kernel<<<(unsigned)((n + 255) / 256), 256, 0, stream>>>(proj_W, prjWt);
  }
  {
    long n = (long)BATCH * TENC * FIN;
    convert_bf16_kernel<<<(unsigned)((n + 255) / 256), 256, 0, stream>>>(frames, frBf, n);
  }
  hipMemsetAsync(cBuf, 0, (size_t)BATCH * HDIM * 4, stream);
  hipMemsetAsync(h0,   0, (size_t)BATCH * HDIM * 2, stream);

  const int GATE_BLOCKS = (4 * 64 * 32) / 256;  // 256 waves -> 32 blocks

  // ---- Encoder: 32 recurrent steps over video frames ----
  for (int t = 0; t < TENC; ++t) {
    const unsigned short* hin  = (t & 1) ? h1 : h0;
    unsigned short*       hout = (t & 1) ? h0 : h1;
    lstm_gates_kernel<<<GATE_BLOCKS, 256, 0, stream>>>(
        frBf + (size_t)t * FIN, TENC * FIN, FIN,
        hin, encWt, KENC, enc_bias, cBuf, hout);
  }
  // t=31 (odd) wrote h0 -> decoder initial hidden state is h0; c continues in cBuf.

  // ---- Decoder: 20 teacher-forced steps; h_t written straight into hs[t] ----
  const unsigned short* hd = h0;
  for (int t = 0; t < LDEC; ++t) {
    embed_gather_kernel<<<(BATCH * WVPAD + 255) / 256, 256, 0, stream>>>(
        caption, t, embeddings, embBf);
    unsigned short* hout = hsBuf + (size_t)t * BATCH * HDIM;
    lstm_gates_kernel<<<GATE_BLOCKS, 256, 0, stream>>>(
        embBf, WVPAD, WVPAD,
        hd, decWt, KDEC, dec_bias, cBuf, hout);
    hd = hout;
  }

  // ---- Projection GEMM: hs @ proj_W + proj_b -> [B, L, 300] fp32 ----
  {
    int waves = ((LDEC * BATCH) / 16) * (NPROJ / 16);  // 160 * 19 = 3040
    int blocks = (waves * 32 + 255) / 256;
    proj_kernel<<<blocks, 256, 0, stream>>>(hsBuf, prjWt, proj_b, out);
  }
}
